// HSTULayer_14834817040550
// MI455X (gfx1250) — compile-verified, matrix operand order checked by reading, not run on hardware
//
#include <hip/hip_runtime.h>
#include <hip/hip_bf16.h>

typedef __bf16 bf16_t;
typedef __attribute__((ext_vector_type(16))) __bf16 v16bf;
typedef __attribute__((ext_vector_type(8)))  __bf16 v8bf;
typedef __attribute__((ext_vector_type(8)))  float  v8f;

#define D_MODEL 1024
#define SEQ     2048
#define BATCH   4
#define NHEAD   16
#define HDIM    64
#define DFF     4096
#define NROWS   (BATCH * SEQ)   // 8192

__device__ __forceinline__ bf16_t f2bf(float f) { return (bf16_t)f; }

// ---------------------------------------------------------------------------
// WMMA A/B fragment loader from a [16-rows x K] row-major tile (wave32).
// lane&15 = row (A: M, B: N when source is pre-transposed); elements 0..7 ->
// K=kbase..kbase+7, 8..15 -> K=kbase+16..kbase+23, kbase = 8*(lane>>4).
// Two 16-byte loads per fragment.
// ---------------------------------------------------------------------------
__device__ __forceinline__ v16bf load_row16(const bf16_t* base, int ld,
                                            int row0, int k0, int lane) {
  int m  = row0 + (lane & 15);
  int kb = k0 + ((lane >> 4) << 3);
  const bf16_t* p = base + (size_t)m * ld + kb;
  v8bf lo = *(const v8bf*)p;          // K = kb .. kb+7
  v8bf hi = *(const v8bf*)(p + 16);   // K = kb+16 .. kb+23
  v16bf f;
#pragma unroll
  for (int i = 0; i < 8; ++i) { f[i] = lo[i]; f[i + 8] = hi[i]; }
  return f;
}

__device__ __forceinline__ v8f wmma_bf16(v16bf a, v16bf b, v8f c) {
  return __builtin_amdgcn_wmma_f32_16x16x32_bf16(false, a, false, b,
                                                 (short)0, c, false, false);
}

// ---------------------------------------------------------------------------
// CDNA5 async global->LDS copy of one 64x32 bf16 weight tile (4 KB).
// 256 threads x 16 B each, tracked by ASYNCcnt.
// ---------------------------------------------------------------------------
__device__ __forceinline__ void async_copy_tile(const bf16_t* Wt, int K,
                                                int col0, int k0,
                                                bf16_t* sB, int t) {
  int r = t >> 2;           // 0..63 : output-column within strip
  int c = (t & 3) << 3;     // 0,8,16,24 : k-offset (bf16 elements)
  unsigned long long g =
      (unsigned long long)(uintptr_t)(Wt + (size_t)(col0 + r) * K + k0 + c);
  unsigned lds = (unsigned)(uintptr_t)(sB + r * 32 + c);  // low 32 bits = LDS offset
  asm volatile("global_load_async_to_lds_b128 %0, %1, off"
               :: "v"(lds), "v"(g) : "memory");
}

// One 16x64 strip worth of WMMAs against a staged 64x32 LDS weight tile.
__device__ __forceinline__ void gemm_step(const bf16_t* A, int K, int row0,
                                          int k0, const bf16_t* sBuf, int lane,
                                          v8f& a0, v8f& a1, v8f& a2, v8f& a3) {
  v16bf a = load_row16(A, K, row0, k0, lane);
  a0 = wmma_bf16(a, load_row16(sBuf, 32,  0, 0, lane), a0);
  a1 = wmma_bf16(a, load_row16(sBuf, 32, 16, 0, lane), a1);
  a2 = wmma_bf16(a, load_row16(sBuf, 32, 32, 0, lane), a2);
  a3 = wmma_bf16(a, load_row16(sBuf, 32, 48, 0, lane), a3);
}

// ---------------------------------------------------------------------------
// f32[R][C] -> bf16 out[C][R] (one-time weight transpose+convert)
// ---------------------------------------------------------------------------
__global__ void cvt_transpose_bf16_kernel(const float* __restrict__ in,
                                          bf16_t* __restrict__ out,
                                          int R, int C) {
  int c = blockIdx.x * 32 + (threadIdx.x & 31);
  int r = blockIdx.y * 8 + (threadIdx.x >> 5);
  if (r < R && c < C) out[(size_t)c * R + r] = f2bf(in[(size_t)r * C + c]);
}

// ---------------------------------------------------------------------------
// LayerNorm: one row (D_MODEL=1024) per block of 256 threads -> bf16 out
// ---------------------------------------------------------------------------
__global__ void layernorm_bf16_kernel(const float* __restrict__ x,
                                      const float* __restrict__ g,
                                      const float* __restrict__ be,
                                      bf16_t* __restrict__ out) {
  __shared__ float ssum[256];
  __shared__ float ssq[256];
  __shared__ float s_mu, s_rstd;
  int r = blockIdx.x, t = threadIdx.x;
  const float* xr = x + (size_t)r * D_MODEL;
  float vals[4], s = 0.f, q = 0.f;
#pragma unroll
  for (int i = 0; i < 4; ++i) {
    float v = xr[t + i * 256];
    vals[i] = v; s += v; q += v * v;
  }
  ssum[t] = s; ssq[t] = q;
  __syncthreads();
  for (int o = 128; o > 0; o >>= 1) {
    if (t < o) { ssum[t] += ssum[t + o]; ssq[t] += ssq[t + o]; }
    __syncthreads();
  }
  if (t == 0) {
    float mu  = ssum[0] * (1.0f / D_MODEL);
    float var = ssq[0] * (1.0f / D_MODEL) - mu * mu;
    s_mu = mu; s_rstd = rsqrtf(var + 1e-5f);
  }
  __syncthreads();
  float mu = s_mu, rstd = s_rstd;
#pragma unroll
  for (int i = 0; i < 4; ++i) {
    int c = t + i * 256;
    out[(size_t)r * D_MODEL + c] = f2bf((vals[i] - mu) * rstd * g[c] + be[c]);
  }
}

// ---------------------------------------------------------------------------
// WMMA GEMM: C[M,N] = A[M,K] * Wt[N,K]^T + bias, optional ReLU / residual.
// Per block: 128 rows x 64 cols. Per wave: one 16x64 strip.
// Weight tiles double-buffered in LDS via async global->LDS, software-
// pipelined with a branch-free steady-state loop (requires K%64==0, K>=128).
// ---------------------------------------------------------------------------
__global__ void gemm_wmma_bf16_kernel(const bf16_t* __restrict__ A,   // [M][K]
                                      const bf16_t* __restrict__ Wt,  // [N][K]
                                      const float* __restrict__ bias,
                                      int M, int N, int K,
                                      float* __restrict__ outf,       // [M][N]
                                      bf16_t* __restrict__ outb,      // [M][N]
                                      bf16_t* __restrict__ outb_t,    // [N][M]
                                      const float* __restrict__ resid,
                                      int do_relu) {
  __shared__ __align__(16) bf16_t sB[2][64 * 32];
  int t    = threadIdx.x;
  int lane = t & 31;
  int wave = t >> 5;
  int row0 = (blockIdx.y * 8 + wave) * 16;
  int col0 = blockIdx.x * 64;

  v8f acc0 = {}, acc1 = {}, acc2 = {}, acc3 = {};

  // prologue: both buffers in flight
  async_copy_tile(Wt, K, col0, 0,  sB[0], t);
  async_copy_tile(Wt, K, col0, 32, sB[1], t);

  // branch-free steady state: 2 k-steps (8 WMMAs) per iteration
  int k0 = 0;
  for (; k0 < K - 64; k0 += 64) {
    asm volatile("s_wait_asynccnt 0x1" ::: "memory");  // buf0's tile landed
    __syncthreads();
    __builtin_prefetch(A + (size_t)(row0 + (lane & 15)) * K + k0 + 64, 0, 3);
    gemm_step(A, K, row0, k0, sB[0], lane, acc0, acc1, acc2, acc3);
    __syncthreads();
    async_copy_tile(Wt, K, col0, k0 + 64, sB[0], t);

    asm volatile("s_wait_asynccnt 0x1" ::: "memory");  // buf1's tile landed
    __syncthreads();
    gemm_step(A, K, row0, k0 + 32, sB[1], lane, acc0, acc1, acc2, acc3);
    __syncthreads();
    async_copy_tile(Wt, K, col0, k0 + 96, sB[1], t);
  }
  // epilogue: last two k-steps, no further issues
  asm volatile("s_wait_asynccnt 0x1" ::: "memory");
  __syncthreads();
  gemm_step(A, K, row0, k0, sB[0], lane, acc0, acc1, acc2, acc3);
  asm volatile("s_wait_asynccnt 0x0" ::: "memory");
  __syncthreads();
  gemm_step(A, K, row0, k0 + 32, sB[1], lane, acc0, acc1, acc2, acc3);

  v8f* accs[4] = { &acc0, &acc1, &acc2, &acc3 };
  int nloc = lane & 15;
#pragma unroll
  for (int nt = 0; nt < 4; ++nt) {
    int n = col0 + nt * 16 + nloc;
    float bv = bias ? bias[n] : 0.f;
#pragma unroll
    for (int e = 0; e < 8; ++e) {
      int m = row0 + e + ((lane >> 4) << 3);
      float v = (*accs[nt])[e] + bv;
      if (do_relu) v = fmaxf(v, 0.f);
      size_t idx = (size_t)m * N + n;
      if (resid) v += resid[idx];
      if (outf) outf[idx] = v;
      if (outb) outb[idx] = f2bf(v);
      if (outb_t) outb_t[(size_t)n * M + m] = f2bf(v);
    }
  }
}

// ---------------------------------------------------------------------------
// Fused sigmoid attention: per wave one 16-query tile of one (b,h).
// Streams 32-key chunks: scores WMMA -> sigmoid -> LDS relayout -> attn@V WMMA.
// V is consumed transposed (Vt[D_MODEL][NROWS]) so fragments are b128 loads.
// x1 = x + attn_out written fused.
// ---------------------------------------------------------------------------
__global__ void attn_sigmoid_wmma_kernel(const bf16_t* __restrict__ Q,
                                         const bf16_t* __restrict__ Kc,
                                         const bf16_t* __restrict__ Vt,
                                         const float* __restrict__ x,
                                         float* __restrict__ x1) {
  const float scale = 0.03125f;  // 1/sqrt(D_MODEL)
  int lane = threadIdx.x & 31;
  int wave = threadIdx.x >> 5;
  int b  = blockIdx.x >> 4;
  int h  = blockIdx.x & 15;
  int q0 = blockIdx.y * 128 + wave * 16;

  const size_t base = (size_t)b * SEQ * D_MODEL + (size_t)h * HDIM;
  const bf16_t* Qb  = Q  + base;
  const bf16_t* Kb  = Kc + base;
  const bf16_t* Vtb = Vt + (size_t)(h * HDIM) * NROWS + (size_t)b * SEQ;

  __shared__ __align__(16) bf16_t probs[8][16 * 32];
  bf16_t* myp = probs[wave];

  // Q tile fragments: hd=64 -> two 32-wide k-steps, reused all chunks
  v16bf qa0 = load_row16(Qb, D_MODEL, q0, 0, lane);
  v16bf qa1 = load_row16(Qb, D_MODEL, q0, 32, lane);

  v8f oacc0 = {}, oacc1 = {}, oacc2 = {}, oacc3 = {};

  for (int kt = 0; kt < SEQ; kt += 32) {
    // scores[m][n] = sum_d q[m][d]*k[n][d]; row-major K as B^T == A-load
    v8f s0 = {}, s1 = {};
    s0 = wmma_bf16(qa0, load_row16(Kb, D_MODEL, kt,      0,  lane), s0);
    s0 = wmma_bf16(qa1, load_row16(Kb, D_MODEL, kt,      32, lane), s0);
    s1 = wmma_bf16(qa0, load_row16(Kb, D_MODEL, kt + 16, 0,  lane), s1);
    s1 = wmma_bf16(qa1, load_row16(Kb, D_MODEL, kt + 16, 32, lane), s1);

    // sigmoid, stage probs (C-layout -> row-major 16x32 in wave-private LDS)
    int n = lane & 15;
#pragma unroll
    for (int e = 0; e < 8; ++e) {
      int m = e + ((lane >> 4) << 3);
      myp[m * 32 + n]      = f2bf(1.0f / (1.0f + __expf(-s0[e] * scale)));
      myp[m * 32 + n + 16] = f2bf(1.0f / (1.0f + __expf(-s1[e] * scale)));
    }
    // re-load probs in A-layout (DS ops are in-order within a wave)
    v16bf pa = load_row16(myp, 32, 0, 0, lane);

    // attn_out += probs(16x32) @ V_chunk(32x64); Vt rows are head-dims
    oacc0 = wmma_bf16(pa, load_row16(Vtb, NROWS,  0, kt, lane), oacc0);
    oacc1 = wmma_bf16(pa, load_row16(Vtb, NROWS, 16, kt, lane), oacc1);
    oacc2 = wmma_bf16(pa, load_row16(Vtb, NROWS, 32, kt, lane), oacc2);
    oacc3 = wmma_bf16(pa, load_row16(Vtb, NROWS, 48, kt, lane), oacc3);
  }

  // x1 = x + attn_out
  v8f* oa[4] = { &oacc0, &oacc1, &oacc2, &oacc3 };
#pragma unroll
  for (int dt = 0; dt < 4; ++dt) {
    int col = h * HDIM + dt * 16 + (lane & 15);
#pragma unroll
    for (int e = 0; e < 8; ++e) {
      int m = q0 + e + ((lane >> 4) << 3);
      size_t idx = ((size_t)b * SEQ + m) * D_MODEL + col;
      x1[idx] = x[idx] + (*oa[dt])[e];
    }
  }
}

// ---------------------------------------------------------------------------
// Host-side orchestration
// ---------------------------------------------------------------------------
extern "C" void kernel_launch(void* const* d_in, const int* in_sizes, int n_in,
                              void* d_out, int out_size, void* d_ws, size_t ws_size,
                              hipStream_t stream) {
  const float* x   = (const float*)d_in[0];
  const float* Wq  = (const float*)d_in[1];
  const float* bq  = (const float*)d_in[2];
  const float* Wk  = (const float*)d_in[3];
  const float* bk  = (const float*)d_in[4];
  const float* Wv  = (const float*)d_in[5];
  const float* bv  = (const float*)d_in[6];
  const float* W1  = (const float*)d_in[7];
  const float* b1  = (const float*)d_in[8];
  const float* W2  = (const float*)d_in[9];
  const float* b2  = (const float*)d_in[10];
  const float* g1  = (const float*)d_in[11];
  const float* be1 = (const float*)d_in[12];
  const float* g2  = (const float*)d_in[13];
  const float* be2 = (const float*)d_in[14];
  float* out = (float*)d_out;

  // workspace carve-up
  bf16_t* wqt  = (bf16_t*)d_ws;                          // [1024][1024]
  bf16_t* wkt  = wqt  + (size_t)D_MODEL * D_MODEL;
  bf16_t* wvt  = wkt  + (size_t)D_MODEL * D_MODEL;
  bf16_t* w1t  = wvt  + (size_t)D_MODEL * D_MODEL;       // [4096][1024]
  bf16_t* w2t  = w1t  + (size_t)D_MODEL * DFF;           // [1024][4096]
  bf16_t* xnb  = w2t  + (size_t)DFF * D_MODEL;           // [8192][1024]
  bf16_t* qb   = xnb  + (size_t)NROWS * D_MODEL;         // [8192][1024]
  bf16_t* kbuf = qb   + (size_t)NROWS * D_MODEL;         // [8192][1024]
  bf16_t* vtb  = kbuf + (size_t)NROWS * D_MODEL;         // [1024][8192] (V^T)
  bf16_t* hb   = vtb  + (size_t)NROWS * D_MODEL;         // [8192][4096]
  bf16_t* xn2b = hb   + (size_t)NROWS * DFF;             // [8192][1024]
  float*  x1   = (float*)(xn2b + (size_t)NROWS * D_MODEL);

  // 1) weights -> bf16, transposed to [N][K]
  cvt_transpose_bf16_kernel<<<dim3(D_MODEL / 32, D_MODEL / 8), 256, 0, stream>>>(Wq, wqt, D_MODEL, D_MODEL);
  cvt_transpose_bf16_kernel<<<dim3(D_MODEL / 32, D_MODEL / 8), 256, 0, stream>>>(Wk, wkt, D_MODEL, D_MODEL);
  cvt_transpose_bf16_kernel<<<dim3(D_MODEL / 32, D_MODEL / 8), 256, 0, stream>>>(Wv, wvt, D_MODEL, D_MODEL);
  cvt_transpose_bf16_kernel<<<dim3(DFF / 32,     D_MODEL / 8), 256, 0, stream>>>(W1, w1t, D_MODEL, DFF);
  cvt_transpose_bf16_kernel<<<dim3(D_MODEL / 32, DFF / 8),     256, 0, stream>>>(W2, w2t, DFF, D_MODEL);

  // 2) LN1
  layernorm_bf16_kernel<<<NROWS, 256, 0, stream>>>(x, g1, be1, xnb);

  // 3) QKV projections (V written transposed for attention)
  {
    dim3 grid(D_MODEL / 64, NROWS / 128);
    gemm_wmma_bf16_kernel<<<grid, 256, 0, stream>>>(xnb, wqt, bq, NROWS, D_MODEL,
                                                    D_MODEL, nullptr, qb, nullptr, nullptr, 0);
    gemm_wmma_bf16_kernel<<<grid, 256, 0, stream>>>(xnb, wkt, bk, NROWS, D_MODEL,
                                                    D_MODEL, nullptr, kbuf, nullptr, nullptr, 0);
    gemm_wmma_bf16_kernel<<<grid, 256, 0, stream>>>(xnb, wvt, bv, NROWS, D_MODEL,
                                                    D_MODEL, nullptr, nullptr, vtb, nullptr, 0);
  }

  // 4) fused sigmoid attention + residual -> x1
  attn_sigmoid_wmma_kernel<<<dim3(BATCH * NHEAD, SEQ / 128), 256, 0, stream>>>(qb, kbuf, vtb, x, x1);

  // 5) LN2
  layernorm_bf16_kernel<<<NROWS, 256, 0, stream>>>(x1, g2, be2, xn2b);

  // 6) FFN1 (ReLU, bf16 out)
  gemm_wmma_bf16_kernel<<<dim3(DFF / 64, NROWS / 128), 256, 0, stream>>>(
      xn2b, w1t, b1, NROWS, DFF, D_MODEL, nullptr, hb, nullptr, nullptr, 1);

  // 7) FFN2 + residual -> out (f32)
  gemm_wmma_bf16_kernel<<<dim3(D_MODEL / 64, NROWS / 128), 256, 0, stream>>>(
      hb, w2t, b2, NROWS, D_MODEL, DFF, out, nullptr, nullptr, x1, 0);
}
